// MambaBlock_65197603553742
// MI455X (gfx1250) — compile-verified, matrix operand order checked by reading, not run on hardware
//
#include <hip/hip_runtime.h>
#include <hip/hip_bf16.h>

// ---------------------------------------------------------------------------
// Mamba block for MI455X (gfx1250, wave32, WMMA + async global->LDS).
//   B=8, L=4096, DIM=64, D_IN=128, N=16, DT_RANK=4, D_CONV=4
// Pipeline:
//   K1  (WMMA f16->f32, async-LDS staged A): xr = x @ in_proj_w^T -> u_pre,res
//   K2  (VALU): depthwise causal conv(4) + bias + SiLU -> u
//   K3  (VALU/LDS): dbl = u @ x_proj_w^T ; delta = softplus(dt_proj) ; B,C
//   K4a (VALU): per-chunk local scans (a-prod, b-scan) over L chunks of 256
//   K4b (VALU): cross-chunk combine -> per-chunk initial states
//   K4c (VALU): re-scan chunks with init state, fuse y = h.C + u*D, *silu(res)
//   K5  (WMMA f16->f32, async-LDS staged A): out = ygated @ out_proj_w^T
// ---------------------------------------------------------------------------

#define B_   8
#define L_   4096
#define DIM_ 64
#define DIN_ 128
#define NS_  16
#define DTR_ 4
#define KW_  4
#define M_   (B_ * L_)      // 32768 GEMM rows
#define NC_  16             // scan chunks
#define CH_  (L_ / NC_)     // 256 steps per chunk

typedef __attribute__((ext_vector_type(16))) _Float16 v16h;
typedef __attribute__((ext_vector_type(8)))  float    v8f;

__device__ __forceinline__ float silu_f(float v) {
  return v / (1.0f + __expf(-v));
}
__device__ __forceinline__ float softplus_f(float v) {
  return (v > 20.0f) ? v : log1pf(__expf(v));
}

// Per-lane K index inside a 16-bit A/B WMMA fragment (16x32 tile, wave32).
__device__ __forceinline__ int wmma_k_of(int i, int lane) {
  return ((i >> 3) << 4) + (i & 7) + ((lane & 16) ? 8 : 0);
}

// CDNA5 async copy: 16 bytes per lane, global -> LDS, tracked by ASYNCcnt.
__device__ __forceinline__ void async_copy_b128(unsigned lds_off,
                                                const float* gaddr) {
  asm volatile("global_load_async_to_lds_b128 %0, %1, off"
               :: "v"(lds_off), "v"(gaddr) : "memory");
}
__device__ __forceinline__ void wait_asynccnt0() {
  asm volatile("s_wait_asynccnt 0" ::: "memory");
}

// ---------------------------------------------------------------------------
// K1: xr[m, j] = sum_k x[m,k] * in_proj_w[j,k];  j<128 -> u_pre, else res
//     M=32768, K=64, N=256. One 16x16 tile per wave; A staged via async-LDS.
// ---------------------------------------------------------------------------
__global__ void k1_in_proj_wmma(const float* __restrict__ x,
                                const float* __restrict__ w,
                                float* __restrict__ u_pre,
                                float* __restrict__ res) {
  __shared__ float sA[8][16][DIM_];                 // 32 KB, wave-private slices
  const int lane = threadIdx.x & 31;
  const int wave = threadIdx.x >> 5;
  const int tile = blockIdx.x * 8 + wave;           // 32768 tiles total
  const int mtile = tile >> 4;                      // /16 ntiles
  const int ntile = tile & 15;
  const int row0 = mtile * 16;

  // Stage A tile (16 rows x 64 f32 = 4 KB) with async b128 copies:
  // 256 chunks of 16B, 8 per lane. Row = chunk/16, 16B-unit col = chunk%16.
#pragma unroll
  for (int i = 0; i < 8; ++i) {
    const int chunk = lane + 32 * i;
    const int r = chunk >> 4;
    const int cc = chunk & 15;
    async_copy_b128((unsigned)(uintptr_t)&sA[wave][r][cc * 4],
                    x + (size_t)(row0 + r) * DIM_ + cc * 4);
  }
  wait_asynccnt0();                                  // wave-private: no barrier

  const int arow = lane & 15;
  const int col = ntile * 16 + (lane & 15);          // B col (N)

  v8f acc = {};
  for (int kc = 0; kc < 2; ++kc) {
    v16h a, b;
    const int k0 = kc * 32;
#pragma unroll
    for (int i = 0; i < 16; ++i) {
      const int k = k0 + wmma_k_of(i, lane);
      a[i] = (_Float16)sA[wave][arow][k];
      b[i] = (_Float16)w[col * DIM_ + k];
    }
    acc = __builtin_amdgcn_wmma_f32_16x16x32_f16(
        false, a, false, b, (short)0, acc, false, false);
  }

#pragma unroll
  for (int r = 0; r < 8; ++r) {
    const int m = row0 + r + ((lane & 16) ? 8 : 0);
    const int n = ntile * 16 + (lane & 15);
    const float v = acc[r];
    if (n < DIN_) u_pre[m * DIN_ + n] = v;
    else          res[m * DIN_ + (n - DIN_)] = v;
  }
}

// ---------------------------------------------------------------------------
// K2: depthwise causal conv over L (width 4) + bias + SiLU.
// ---------------------------------------------------------------------------
__global__ void k2_conv_silu(const float* __restrict__ u_pre,
                             const float* __restrict__ conv_w,
                             const float* __restrict__ conv_b,
                             float* __restrict__ u) {
  const int idx = blockIdx.x * blockDim.x + threadIdx.x;
  if (idx >= B_ * L_ * DIN_) return;
  const int d = idx % DIN_;
  const int l = (idx / DIN_) % L_;
  const int b = idx / (DIN_ * L_);
  float s = conv_b[d];
#pragma unroll
  for (int k = 0; k < KW_; ++k) {
    const int ll = l - (KW_ - 1) + k;
    if (ll >= 0) s += conv_w[d * KW_ + k] * u_pre[(b * L_ + ll) * DIN_ + d];
  }
  u[idx] = silu_f(s);
}

// ---------------------------------------------------------------------------
// K3: per row m: dbl = x_proj_w(36x128) @ u_row; split delta_r/B/C;
//     delta = softplus(dt_proj_w @ delta_r + dt_proj_b). Block = 128 threads.
// ---------------------------------------------------------------------------
__global__ void k3_proj_delta(const float* __restrict__ u,
                              const float* __restrict__ x_proj_w,
                              const float* __restrict__ dt_proj_w,
                              const float* __restrict__ dt_proj_b,
                              float* __restrict__ Bm,
                              float* __restrict__ Cm,
                              float* __restrict__ delta) {
  __shared__ float urow[DIN_];
  __shared__ float dblr[DTR_ + 2 * NS_];
  const int m = blockIdx.x;
  const int t = threadIdx.x;
  urow[t] = u[m * DIN_ + t];
  __syncthreads();

  if (t < DTR_ + 2 * NS_) {
    float s = 0.0f;
#pragma unroll 8
    for (int d = 0; d < DIN_; ++d) s += x_proj_w[t * DIN_ + d] * urow[d];
    dblr[t] = s;
    if (t >= DTR_ && t < DTR_ + NS_)       Bm[m * NS_ + (t - DTR_)] = s;
    else if (t >= DTR_ + NS_)              Cm[m * NS_ + (t - DTR_ - NS_)] = s;
  }
  __syncthreads();

  float s = dt_proj_b[t];
#pragma unroll
  for (int r = 0; r < DTR_; ++r) s += dt_proj_w[t * DTR_ + r] * dblr[r];
  delta[m * DIN_ + t] = softplus_f(s);
}

// ---------------------------------------------------------------------------
// K4a: per-(b,d,n,chunk) local scan: (aprod, bscan) over 256 steps.
// ---------------------------------------------------------------------------
__global__ void k4a_chunk_scan(const float* __restrict__ delta,
                               const float* __restrict__ u,
                               const float* __restrict__ Bm,
                               const float* __restrict__ A_log,
                               float* __restrict__ chA,
                               float* __restrict__ chB) {
  const int idx = blockIdx.x * blockDim.x + threadIdx.x;
  if (idx >= B_ * DIN_ * NC_ * NS_) return;
  const int n = idx % NS_;
  const int c = (idx / NS_) % NC_;
  const int d = (idx / (NS_ * NC_)) % DIN_;
  const int b = idx / (NS_ * NC_ * DIN_);

  const float A = -__expf(A_log[d * NS_ + n]);
  float ap = 1.0f, bp = 0.0f;
  const int l0 = c * CH_;
  for (int l = l0; l < l0 + CH_; ++l) {
    const float dt = delta[(b * L_ + l) * DIN_ + d];
    const float uu = u[(b * L_ + l) * DIN_ + d];
    const float bb = Bm[(b * L_ + l) * NS_ + n];
    const float da = __expf(dt * A);
    ap *= da;
    bp = da * bp + dt * uu * bb;
  }
  chA[idx] = ap;
  chB[idx] = bp;
}

// ---------------------------------------------------------------------------
// K4b: cross-chunk sequential combine (16 steps) -> per-chunk initial h.
// ---------------------------------------------------------------------------
__global__ void k4b_combine(const float* __restrict__ chA,
                            const float* __restrict__ chB,
                            float* __restrict__ hinit) {
  const int idx = blockIdx.x * blockDim.x + threadIdx.x;
  if (idx >= B_ * DIN_ * NS_) return;
  const int n = idx % NS_;
  const int bd = idx / NS_;   // b*DIN + d
  float h = 0.0f;
#pragma unroll
  for (int c = 0; c < NC_; ++c) {
    const int j = (bd * NC_ + c) * NS_ + n;
    hinit[j] = h;
    h = chA[j] * h + chB[j];
  }
}

// ---------------------------------------------------------------------------
// K4c: re-scan each chunk with correct init; fuse y = sum_n h*C + u*D and
//      gating y *= silu(res).
// ---------------------------------------------------------------------------
__global__ void k4c_rescan_y(const float* __restrict__ delta,
                             const float* __restrict__ u,
                             const float* __restrict__ Bm,
                             const float* __restrict__ Cm,
                             const float* __restrict__ res,
                             const float* __restrict__ A_log,
                             const float* __restrict__ Dv,
                             const float* __restrict__ hinit,
                             float* __restrict__ ygated) {
  const int idx = blockIdx.x * blockDim.x + threadIdx.x;
  if (idx >= B_ * NC_ * DIN_) return;
  const int d = idx % DIN_;
  const int c = (idx / DIN_) % NC_;
  const int b = idx / (DIN_ * NC_);
  const int bd = b * DIN_ + d;

  float h[NS_], A[NS_];
#pragma unroll
  for (int n = 0; n < NS_; ++n) {
    h[n] = hinit[(bd * NC_ + c) * NS_ + n];
    A[n] = -__expf(A_log[d * NS_ + n]);
  }
  const float Dd = Dv[d];

  const int l0 = c * CH_;
  for (int l = l0; l < l0 + CH_; ++l) {
    const float dt = delta[(b * L_ + l) * DIN_ + d];
    const float uu = u[(b * L_ + l) * DIN_ + d];
    const float du = dt * uu;
    float y = 0.0f;
#pragma unroll
    for (int n = 0; n < NS_; ++n) {
      const float da = __expf(dt * A[n]);
      h[n] = da * h[n] + du * Bm[(b * L_ + l) * NS_ + n];
      y += h[n] * Cm[(b * L_ + l) * NS_ + n];
    }
    y += uu * Dd;
    const float r = res[(b * L_ + l) * DIN_ + d];
    ygated[(b * L_ + l) * DIN_ + d] = y * silu_f(r);
  }
}

// ---------------------------------------------------------------------------
// K5: out[m, o] = sum_d ygated[m,d] * out_proj_w[o,d]. M=32768, K=128, N=64.
//     128-thread blocks (4 waves); A tile staged via async-LDS (8 KB/wave).
// ---------------------------------------------------------------------------
__global__ void k5_out_proj_wmma(const float* __restrict__ yg,
                                 const float* __restrict__ w,
                                 float* __restrict__ out) {
  __shared__ float sA[4][16][DIN_];                 // 32 KB, wave-private
  const int lane = threadIdx.x & 31;
  const int wave = threadIdx.x >> 5;
  const int tile = blockIdx.x * 4 + wave;           // 2048*4 = 8192 tiles
  const int mtile = tile >> 2;                      // /4 ntiles
  const int ntile = tile & 3;
  const int row0 = mtile * 16;

  // Stage A tile (16 rows x 128 f32 = 8 KB): 512 x 16B chunks, 16 per lane.
#pragma unroll
  for (int i = 0; i < 16; ++i) {
    const int chunk = lane + 32 * i;
    const int r = chunk >> 5;                       // 32 x 16B chunks per row
    const int cc = chunk & 31;
    async_copy_b128((unsigned)(uintptr_t)&sA[wave][r][cc * 4],
                    yg + (size_t)(row0 + r) * DIN_ + cc * 4);
  }
  wait_asynccnt0();

  const int arow = lane & 15;
  const int col = ntile * 16 + (lane & 15);

  v8f acc = {};
  for (int kc = 0; kc < 4; ++kc) {
    v16h a, b;
    const int k0 = kc * 32;
#pragma unroll
    for (int i = 0; i < 16; ++i) {
      const int k = k0 + wmma_k_of(i, lane);
      a[i] = (_Float16)sA[wave][arow][k];
      b[i] = (_Float16)w[col * DIN_ + k];
    }
    acc = __builtin_amdgcn_wmma_f32_16x16x32_f16(
        false, a, false, b, (short)0, acc, false, false);
  }

#pragma unroll
  for (int r = 0; r < 8; ++r) {
    const int m = row0 + r + ((lane & 16) ? 8 : 0);
    const int n = ntile * 16 + (lane & 15);
    out[m * DIM_ + n] = acc[r];
  }
}

// ---------------------------------------------------------------------------
extern "C" void kernel_launch(void* const* d_in, const int* in_sizes, int n_in,
                              void* d_out, int out_size, void* d_ws, size_t ws_size,
                              hipStream_t stream) {
  const float* x         = (const float*)d_in[0];
  // d_in[1] = latent (unused)
  const float* in_proj_w = (const float*)d_in[2];
  const float* conv_w    = (const float*)d_in[3];
  const float* conv_b    = (const float*)d_in[4];
  const float* x_proj_w  = (const float*)d_in[5];
  const float* dt_proj_w = (const float*)d_in[6];
  const float* dt_proj_b = (const float*)d_in[7];
  const float* out_proj_w= (const float*)d_in[8];
  const float* A_log     = (const float*)d_in[9];
  const float* Dv        = (const float*)d_in[10];
  float* out = (float*)d_out;

  // workspace layout (floats), ~91 MB total
  float* ws = (float*)d_ws;
  const size_t SZ_BLD = (size_t)B_ * L_ * DIN_;        // 4194304
  const size_t SZ_BLN = (size_t)B_ * L_ * NS_;         //  524288
  const size_t SZ_CH  = (size_t)B_ * DIN_ * NC_ * NS_; //  262144
  float* u_pre = ws;                 ws += SZ_BLD;
  float* res   = ws;                 ws += SZ_BLD;
  float* u     = ws;                 ws += SZ_BLD;
  float* delta = ws;                 ws += SZ_BLD;
  float* Bm    = ws;                 ws += SZ_BLN;
  float* Cm    = ws;                 ws += SZ_BLN;
  float* chA   = ws;                 ws += SZ_CH;
  float* chB   = ws;                 ws += SZ_CH;
  float* hinit = ws;                 ws += SZ_CH;
  float* yg    = ws;                 ws += SZ_BLD;

  k1_in_proj_wmma<<<4096, 256, 0, stream>>>(x, in_proj_w, u_pre, res);

  k2_conv_silu<<<(B_ * L_ * DIN_ + 255) / 256, 256, 0, stream>>>(
      u_pre, conv_w, conv_b, u);

  k3_proj_delta<<<M_, DIN_, 0, stream>>>(
      u, x_proj_w, dt_proj_w, dt_proj_b, Bm, Cm, delta);

  k4a_chunk_scan<<<(B_ * DIN_ * NC_ * NS_ + 255) / 256, 256, 0, stream>>>(
      delta, u, Bm, A_log, chA, chB);
  k4b_combine<<<(B_ * DIN_ * NS_ + 255) / 256, 256, 0, stream>>>(
      chA, chB, hinit);
  k4c_rescan_y<<<(B_ * NC_ * DIN_ + 255) / 256, 256, 0, stream>>>(
      delta, u, Bm, Cm, res, A_log, Dv, hinit, yg);

  k5_out_proj_wmma<<<2048, 128, 0, stream>>>(yg, out_proj_w, out);

  (void)in_sizes; (void)n_in; (void)out_size; (void)ws_size;
}
